// ATT_model_51350628991054
// MI455X (gfx1250) — matrix operand order repaired and checked by reading, not compiled, not run on hardware
//
#include <hip/hip_runtime.h>
#include <hip/hip_bf16.h>

// ---------------------------------------------------------------------------
// Attention-Sum reader on MI455X (gfx1250, wave32, WMMA f16 16x16x32).
// Pipeline: convert weights f32->f16 | embed-gather->f16 | xw GEMM (WMMA)
//           | BiGRU scan (WMMA per step, f32 gates) | scores GEMM (WMMA)
//           | fused masked softmax / beta-average / probs | segment argmax.
// ---------------------------------------------------------------------------

#define CV 50000
#define CE 384
#define CH 256
#define CG 768    // 3*CH
#define CB 64
#define CTD 1024
#define CTQ 32

typedef __attribute__((ext_vector_type(16))) _Float16 v16h;
typedef __attribute__((ext_vector_type(8)))  _Float16 v8h;
typedef __attribute__((ext_vector_type(8)))  float    v8f;

union AFrag { v16h v; v8h h[2]; };

__device__ __forceinline__ v8f wmma16(v16h a, v16h b, v8f c) {
  // D = A(16x32 f16) * B(32x16 f16) + C(16x16 f32)
  return __builtin_amdgcn_wmma_f32_16x16x32_f16(
      false, a, false, b, (short)0, c, false, false);
}

// A fragment: row-major A, lda in elements. Caller pre-offsets base by k-chunk.
// ISA 16-bit A 16x32 layout: lane(0-15): K[0..8)+K[16..24); lane(16-31): +8.
__device__ __forceinline__ v16h load_a16(const _Float16* base, int lda, int lane) {
  const int row = lane & 15;
  const int hi  = (lane >> 4) & 1;
  AFrag f;
  f.h[0] = *(const v8h*)(base + (size_t)row * lda + hi * 8);
  f.h[1] = *(const v8h*)(base + (size_t)row * lda + hi * 8 + 16);
  return f.v;
}

// B fragment where B = W^T and W is stored row-major (N x K): B[k][n] = W[n][k].
// ISA 16-bit B 32x16 layout: lane n = lane&15, K[0..8)+K[16..24) for lanes 0-15; +8 for lanes 16-31
// -> two 16B loads per lane.
__device__ __forceinline__ v16h load_bT(const _Float16* W, int ldw, int n0, int k0, int lane) {
  const int n  = n0 + (lane & 15);
  const int hi = (lane >> 4) & 1;
  AFrag f;
  f.h[0] = *(const v8h*)(W + (size_t)n * ldw + k0 + hi * 8);
  f.h[1] = *(const v8h*)(W + (size_t)n * ldw + k0 + hi * 8 + 16);
  return f.v;
}

// ---------------------------------------------------------------------------
__global__ void k_zero(float* p, int n) {
  int i = blockIdx.x * blockDim.x + threadIdx.x;
  const int stride = gridDim.x * blockDim.x;
  for (; i < n; i += stride) p[i] = 0.0f;
}

__global__ void k_f32_to_f16(const float* __restrict__ src, _Float16* __restrict__ dst, int n) {
  int i = blockIdx.x * blockDim.x + threadIdx.x;
  const int stride = gridDim.x * blockDim.x;
  for (; i < n; i += stride) dst[i] = (_Float16)src[i];
}

// one block per token: out[t][0..E) = (f16)emb[tok][0..E)
__global__ void k_gather(const int* __restrict__ toks, const float* __restrict__ emb,
                         _Float16* __restrict__ out) {
  const int t = blockIdx.x;
  const int tok = toks[t];
  const float* src = emb + (size_t)tok * CE;
  _Float16* dst = out + (size_t)t * CE;
  for (int e = threadIdx.x; e < CE; e += blockDim.x) dst[e] = (_Float16)src[e];
}

// ---------------------------------------------------------------------------
// xw = X @ Wih^T + bih  (M x 768, K=384), f16 in, f32 accum, f16 out.
// grid: (M/16, 6, 2 dirs), block 256 (8 waves, 16 cols each).
__global__ void __launch_bounds__(256)
k_xw_gemm(const _Float16* __restrict__ X,
          const _Float16* __restrict__ WihF, const _Float16* __restrict__ WihB,
          const float* __restrict__ bihF, const float* __restrict__ bihB,
          _Float16* __restrict__ xwF, _Float16* __restrict__ xwB) {
  const int dir = blockIdx.z;
  const _Float16* W   = dir ? WihB : WihF;
  const float*    bia = dir ? bihB : bihF;
  _Float16*       xw  = dir ? xwB  : xwF;

  const int m0   = blockIdx.x * 16;
  const int wave = threadIdx.x >> 5;
  const int lane = threadIdx.x & 31;
  const int n0   = blockIdx.y * 128 + wave * 16;

  const _Float16* A = X + (size_t)m0 * CE;
  v8f c = {0.f, 0.f, 0.f, 0.f, 0.f, 0.f, 0.f, 0.f};
#pragma unroll
  for (int kk = 0; kk < 12; ++kk) {
    v16h a = load_a16(A + kk * 32, CE, lane);
    v16h b = load_bT(W, CE, n0, kk * 32, lane);
    c = wmma16(a, b, c);
  }
  const int col = lane & 15, hi = lane >> 4;
  const float bn = bia[n0 + col];
#pragma unroll
  for (int v = 0; v < 8; ++v) {
    const int m = hi * 8 + v;
    xw[(size_t)(m0 + m) * CG + n0 + col] = (_Float16)(c[v] + bn);
  }
}

// ---------------------------------------------------------------------------
// GRU scan: one WG handles 16 batch rows x 1 direction, T sequential steps.
// Per step: gh = h @ Whh^T via WMMA (wave owns matched r/z/n gate columns,
// 32 hidden units), then GRU elementwise in registers.
__global__ void __launch_bounds__(256)
k_gru_scan(const _Float16* __restrict__ xwF, const _Float16* __restrict__ xwB,
           const _Float16* __restrict__ WhhF, const _Float16* __restrict__ WhhB,
           const float* __restrict__ bhhF, const float* __restrict__ bhhB,
           const int* __restrict__ lens, int T, _Float16* __restrict__ outH) {
  const int dir = blockIdx.y;
  const int b0  = blockIdx.x * 16;
  const _Float16* xw  = dir ? xwB  : xwF;
  const _Float16* Whh = dir ? WhhB : WhhF;
  const float*    bhh = dir ? bhhB : bhhF;

  __shared__ __align__(16) float    sh_h[16][CH];        // carried state, f32
  __shared__ __align__(16) _Float16 sh_h16[16][CH + 8];  // A staging, f16, padded

  const int tid  = threadIdx.x;
  const int wave = tid >> 5;
  const int lane = tid & 31;
  const int col  = lane & 15;
  const int hi   = lane >> 4;
  const int u0   = wave * 32;  // this wave's 32 hidden units

  for (int i = tid; i < 16 * CH; i += 256) { sh_h[i >> 8][i & 255] = 0.0f; }
  for (int i = tid; i < 16 * (CH + 8); i += 256) { (&sh_h16[0][0])[i] = (_Float16)0.0f; }
  __syncthreads();

  int mylen[8];
#pragma unroll
  for (int v = 0; v < 8; ++v) mylen[v] = lens[b0 + hi * 8 + v];

  const float bh_r0 = bhh[u0 + col],            bh_r1 = bhh[u0 + 16 + col];
  const float bh_z0 = bhh[CH + u0 + col],       bh_z1 = bhh[CH + u0 + 16 + col];
  const float bh_n0 = bhh[2 * CH + u0 + col],   bh_n1 = bhh[2 * CH + u0 + 16 + col];

  for (int t = 0; t < T; ++t) {
    v8f cR0 = {0.f,0.f,0.f,0.f,0.f,0.f,0.f,0.f}, cR1 = cR0;
    v8f cZ0 = cR0, cZ1 = cR0, cN0 = cR0, cN1 = cR0;
#pragma unroll
    for (int kk = 0; kk < 8; ++kk) {
      v16h a = load_a16(&sh_h16[0][0] + kk * 32, CH + 8, lane);
      cR0 = wmma16(a, load_bT(Whh, CH, u0,               kk * 32, lane), cR0);
      cR1 = wmma16(a, load_bT(Whh, CH, u0 + 16,          kk * 32, lane), cR1);
      cZ0 = wmma16(a, load_bT(Whh, CH, CH + u0,          kk * 32, lane), cZ0);
      cZ1 = wmma16(a, load_bT(Whh, CH, CH + u0 + 16,     kk * 32, lane), cZ1);
      cN0 = wmma16(a, load_bT(Whh, CH, 2 * CH + u0,      kk * 32, lane), cN0);
      cN1 = wmma16(a, load_bT(Whh, CH, 2 * CH + u0 + 16, kk * 32, lane), cN1);
    }
    __syncthreads();  // all frag reads of sh_h16 done before we overwrite it

#pragma unroll
    for (int v = 0; v < 8; ++v) {
      const int m = hi * 8 + v;
      const int b = b0 + m;
      const int len = mylen[v];
      const int tt = dir ? ((t < len) ? (len - 1 - t) : 0) : t;
      const _Float16* gi = xw + ((size_t)b * T + tt) * CG;
      const float crv[2] = {cR0[v], cR1[v]};
      const float czv[2] = {cZ0[v], cZ1[v]};
      const float cnv[2] = {cN0[v], cN1[v]};
      const float brv[2] = {bh_r0, bh_r1};
      const float bzv[2] = {bh_z0, bh_z1};
      const float bnv[2] = {bh_n0, bh_n1};
#pragma unroll
      for (int s2 = 0; s2 < 2; ++s2) {
        const int u = u0 + s2 * 16 + col;
        const float gr = (float)gi[u]          + crv[s2] + brv[s2];
        const float gz = (float)gi[CH + u]     + czv[s2] + bzv[s2];
        const float gn = (float)gi[2 * CH + u];
        const float hn = cnv[s2] + bnv[s2];
        const float rg = 1.0f / (1.0f + expf(-gr));
        const float zg = 1.0f / (1.0f + expf(-gz));
        const float ng = tanhf(gn + rg * hn);
        const float ho = sh_h[m][u];
        const float hv = (1.0f - zg) * ng + zg * ho;
        sh_h[m][u] = hv;
        sh_h16[m][u] = (_Float16)hv;
        if (t < len) {
          const int pos = dir ? (len - 1 - t) : t;
          outH[((size_t)b * T + pos) * (2 * CH) + (size_t)dir * CH + u] = (_Float16)hv;
        }
      }
    }
    __syncthreads();  // sh_h16 updated before next step's fragments
  }
}

// ---------------------------------------------------------------------------
// scores[b,d,q] = docH[b,d,:] . qryH[b,q,:]  (K=512). grid (B, TD/16), block 64.
__global__ void __launch_bounds__(64)
k_scores(const _Float16* __restrict__ docH, const _Float16* __restrict__ qryH,
         float* __restrict__ scores) {
  const int b  = blockIdx.x;
  const int d0 = blockIdx.y * 16;
  const int wave = threadIdx.x >> 5;
  const int lane = threadIdx.x & 31;
  const int n0 = wave * 16;

  const _Float16* A = docH + ((size_t)b * CTD + d0) * (2 * CH);
  const _Float16* Q = qryH + (size_t)b * CTQ * (2 * CH);
  v8f c = {0.f, 0.f, 0.f, 0.f, 0.f, 0.f, 0.f, 0.f};
#pragma unroll
  for (int kk = 0; kk < 16; ++kk) {
    v16h a = load_a16(A + kk * 32, 2 * CH, lane);
    v16h bb = load_bT(Q, 2 * CH, n0, kk * 32, lane);
    c = wmma16(a, bb, c);
  }
  const int col = lane & 15, hi = lane >> 4;
#pragma unroll
  for (int v = 0; v < 8; ++v)
    scores[((size_t)b * CTD + d0 + hi * 8 + v) * CTQ + n0 + col] = c[v];
}

// ---------------------------------------------------------------------------
// Fused: masked softmaxes, beta-average, s[d], probs, segment scatter.
__global__ void __launch_bounds__(256)
k_attn_reduce(const float* __restrict__ scores, const int* __restrict__ doc,
              const int* __restrict__ dlens, const int* __restrict__ qlens,
              const int* __restrict__ answers,
              float* __restrict__ sums, float* __restrict__ cnts,
              float* __restrict__ out_probs) {
  const int b = blockIdx.x, tid = threadIdx.x;
  const int dlen = dlens[b], qlen = qlens[b];
  const float* S = scores + (size_t)b * CTD * CTQ;

  __shared__ float red[256][33];
  __shared__ float cm[CTQ], cs[CTQ], bav[CTQ];
  __shared__ float rs[256];

  // pass 1: column max over d (unmasked, matches reference shift)
  float lm[CTQ];
#pragma unroll
  for (int q = 0; q < CTQ; ++q) lm[q] = -3.4e38f;
  for (int d = tid; d < CTD; d += 256) {
    const float* row = S + (size_t)d * CTQ;
#pragma unroll
    for (int q = 0; q < CTQ; ++q) lm[q] = fmaxf(lm[q], row[q]);
  }
#pragma unroll
  for (int q = 0; q < CTQ; ++q) red[tid][q] = lm[q];
  __syncthreads();
  if (tid < CTQ) {
    float m = -3.4e38f;
    for (int i = 0; i < 256; ++i) m = fmaxf(m, red[i][tid]);
    cm[tid] = m;
  }
  __syncthreads();

  // pass 2: alpha denominators (colsum) and beta column sums
  float ls[CTQ], lb[CTQ];
#pragma unroll
  for (int q = 0; q < CTQ; ++q) { ls[q] = 0.0f; lb[q] = 0.0f; }
  for (int d = tid; d < dlen; d += 256) {
    const float* row = S + (size_t)d * CTQ;
    float rmax = -3.4e38f;
#pragma unroll
    for (int q = 0; q < CTQ; ++q) rmax = fmaxf(rmax, row[q]);  // unmasked shift
    float ex[CTQ];
    float rsum = 0.0f;
#pragma unroll
    for (int q = 0; q < CTQ; ++q) {
      const float e = (q < qlen) ? expf(row[q] - rmax) : 0.0f;
      ex[q] = e; rsum += e;
    }
    const float inv = 1.0f / (rsum + 1e-12f);
#pragma unroll
    for (int q = 0; q < CTQ; ++q) {
      if (q < qlen) {
        ls[q] += expf(row[q] - cm[q]);
        lb[q] += ex[q] * inv;
      }
    }
  }
#pragma unroll
  for (int q = 0; q < CTQ; ++q) red[tid][q] = ls[q];
  __syncthreads();
  if (tid < CTQ) { float s = 0.0f; for (int i = 0; i < 256; ++i) s += red[i][tid]; cs[tid] = s; }
  __syncthreads();
#pragma unroll
  for (int q = 0; q < CTQ; ++q) red[tid][q] = lb[q];
  __syncthreads();
  if (tid < CTQ) { float s = 0.0f; for (int i = 0; i < 256; ++i) s += red[i][tid]; bav[tid] = s / (float)dlen; }
  __syncthreads();

  // pass 3: s[d] = sum_q alpha[d,q]*beta_aver[q]; probs; segment scatter
  const int ans = answers[b];
  float pacc = 0.0f;
  for (int d = tid; d < dlen; d += 256) {
    const float* row = S + (size_t)d * CTQ;
    float sd = 0.0f;
    for (int q = 0; q < qlen; ++q)
      sd += expf(row[q] - cm[q]) / (cs[q] + 1e-12f) * bav[q];
    const int tok = doc[b * CTD + d];
    atomicAdd(&sums[(size_t)b * CV + tok], sd);
    atomicAdd(&cnts[(size_t)b * CV + tok], 1.0f);
    if (tok == ans) pacc += sd;
  }
  rs[tid] = pacc;
  __syncthreads();
  for (int off = 128; off > 0; off >>= 1) {
    if (tid < off) rs[tid] += rs[tid + off];
    __syncthreads();
  }
  if (tid == 0) out_probs[b] = rs[0];
}

// ---------------------------------------------------------------------------
// argmax over word_scores = cnts>0 ? sums : -1e30, first-index tie-break.
__global__ void __launch_bounds__(256)
k_argmax(const float* __restrict__ sums, const float* __restrict__ cnts,
         float* __restrict__ out) {
  const int b = blockIdx.x, tid = threadIdx.x;
  const float* s = sums + (size_t)b * CV;
  const float* c = cnts + (size_t)b * CV;
  float best = -3.4e38f;
  int bi = 0x7fffffff;
  for (int v = tid; v < CV; v += 256) {
    const float val = (c[v] > 0.0f) ? s[v] : -1e30f;
    if (val > best || (val == best && v < bi)) { best = val; bi = v; }
  }
  __shared__ float bv[256];
  __shared__ int bx[256];
  bv[tid] = best; bx[tid] = bi;
  __syncthreads();
  for (int off = 128; off > 0; off >>= 1) {
    if (tid < off) {
      const float v2 = bv[tid + off];
      const int i2 = bx[tid + off];
      if (v2 > bv[tid] || (v2 == bv[tid] && i2 < bx[tid])) { bv[tid] = v2; bx[tid] = i2; }
    }
    __syncthreads();
  }
  if (tid == 0) out[CB + b] = (float)bx[0];
}

// ---------------------------------------------------------------------------
extern "C" void kernel_launch(void* const* d_in, const int* in_sizes, int n_in,
                              void* d_out, int out_size, void* d_ws, size_t ws_size,
                              hipStream_t stream) {
  (void)in_sizes; (void)n_in; (void)out_size; (void)ws_size;
  const int*   documents  = (const int*)d_in[0];
  const int*   doc_lens   = (const int*)d_in[1];
  const int*   querys     = (const int*)d_in[2];
  const int*   query_lens = (const int*)d_in[3];
  const int*   answers    = (const int*)d_in[4];
  const float* emb        = (const float*)d_in[5];
  const float* Wih_f      = (const float*)d_in[6];
  const float* Whh_f      = (const float*)d_in[7];
  const float* bih_f      = (const float*)d_in[8];
  const float* bhh_f      = (const float*)d_in[9];
  const float* Wih_b      = (const float*)d_in[10];
  const float* Whh_b      = (const float*)d_in[11];
  const float* bih_b      = (const float*)d_in[12];
  const float* bhh_b      = (const float*)d_in[13];
  float* out = (float*)d_out;

  char* ws = (char*)d_ws;
  size_t off = 0;
  auto alloc = [&](size_t bytes) -> char* {
    char* p = ws + off;
    off = (off + bytes + 255) & ~(size_t)255;
    return p;
  };

  _Float16* wihF16 = (_Float16*)alloc((size_t)CG * CE * 2);
  _Float16* wihB16 = (_Float16*)alloc((size_t)CG * CE * 2);
  _Float16* whhF16 = (_Float16*)alloc((size_t)CG * CH * 2);
  _Float16* whhB16 = (_Float16*)alloc((size_t)CG * CH * 2);
  _Float16* doc_xh = (_Float16*)alloc((size_t)CB * CTD * CE * 2);
  _Float16* qry_xh = (_Float16*)alloc((size_t)CB * CTQ * CE * 2);
  _Float16* xw_docF = (_Float16*)alloc((size_t)CB * CTD * CG * 2);
  _Float16* xw_docB = (_Float16*)alloc((size_t)CB * CTD * CG * 2);
  _Float16* xw_qryF = (_Float16*)alloc((size_t)CB * CTQ * CG * 2);
  _Float16* xw_qryB = (_Float16*)alloc((size_t)CB * CTQ * CG * 2);
  _Float16* docH = (_Float16*)alloc((size_t)CB * CTD * 2 * CH * 2);
  _Float16* qryH = (_Float16*)alloc((size_t)CB * CTQ * 2 * CH * 2);
  float* scores = (float*)alloc((size_t)CB * CTD * CTQ * 4);
  float* sums   = (float*)alloc((size_t)CB * CV * 4);
  float* cnts   = (float*)alloc((size_t)CB * CV * 4);

  // weight conversion
  k_f32_to_f16<<<dim3(1152), dim3(256), 0, stream>>>(Wih_f, wihF16, CG * CE);
  k_f32_to_f16<<<dim3(1152), dim3(256), 0, stream>>>(Wih_b, wihB16, CG * CE);
  k_f32_to_f16<<<dim3(768),  dim3(256), 0, stream>>>(Whh_f, whhF16, CG * CH);
  k_f32_to_f16<<<dim3(768),  dim3(256), 0, stream>>>(Whh_b, whhB16, CG * CH);

  // zero outputs that are sparsely written (valid positions only) + segment bufs
  k_zero<<<dim3(4096), dim3(256), 0, stream>>>((float*)docH, CB * CTD * CH);       // f16 count/2
  k_zero<<<dim3(2048), dim3(256), 0, stream>>>((float*)qryH, CB * CTQ * CH);       // f16 count/2
  k_zero<<<dim3(4096), dim3(256), 0, stream>>>(sums, CB * CV);
  k_zero<<<dim3(4096), dim3(256), 0, stream>>>(cnts, CB * CV);

  // embedding gather -> f16
  k_gather<<<dim3(CB * CTD), dim3(128), 0, stream>>>(documents, emb, doc_xh);
  k_gather<<<dim3(CB * CTQ), dim3(128), 0, stream>>>(querys, emb, qry_xh);

  // xw = x @ Wih^T + bih, both directions
  k_xw_gemm<<<dim3(CB * CTD / 16, 6, 2), dim3(256), 0, stream>>>(
      doc_xh, wihF16, wihB16, bih_f, bih_b, xw_docF, xw_docB);
  k_xw_gemm<<<dim3(CB * CTQ / 16, 6, 2), dim3(256), 0, stream>>>(
      qry_xh, wihF16, wihB16, bih_f, bih_b, xw_qryF, xw_qryB);

  // BiGRU scans
  k_gru_scan<<<dim3(CB / 16, 2), dim3(256), 0, stream>>>(
      xw_docF, xw_docB, whhF16, whhB16, bhh_f, bhh_b, doc_lens, CTD, docH);
  k_gru_scan<<<dim3(CB / 16, 2), dim3(256), 0, stream>>>(
      xw_qryF, xw_qryB, whhF16, whhB16, bhh_f, bhh_b, query_lens, CTQ, qryH);

  // attention scores
  k_scores<<<dim3(CB, CTD / 16), dim3(64), 0, stream>>>(docH, qryH, scores);

  // fused softmaxes + probs + segment scatter
  k_attn_reduce<<<dim3(CB), dim3(256), 0, stream>>>(
      scores, documents, doc_lens, query_lens, answers, sums, cnts, out);

  // per-batch argmax over vocabulary
  k_argmax<<<dim3(CB), dim3(256), 0, stream>>>(sums, cnts, out);
}
